// MultiHeadAttention_71133248356429
// MI455X (gfx1250) — matrix operand order repaired and checked by reading, not compiled, not run on hardware
//
#include <hip/hip_runtime.h>
#include <hip/hip_bf16.h>

// ---------------------------------------------------------------------------
// MHA forward (RoPE + causal softmax) for gfx1250 (MI455X), wave32 WMMA.
//   D_MODEL=1024, H=16, d=64, B=4, S=2048
// Pipeline:
//   k0: precompute RoPE (cos,sin) table [2048 x 32]          -> csTab (f32x2)
//   k1: KQV = X @ Wkqv^T (f16 WMMA, f32 acc) + fused RoPE    -> Qh/Kh/Vh (f16)
//   k2: flash attention (transposed scores), causal          -> acts (f16)
//   k3: out = acts @ Wo^T (f16 WMMA, f32 acc)                -> d_out (f32)
// ---------------------------------------------------------------------------

typedef __attribute__((ext_vector_type(16))) _Float16 v16h;
typedef __attribute__((ext_vector_type(8)))  _Float16 v8h;
typedef __attribute__((ext_vector_type(8)))  float    v8f;
typedef __attribute__((ext_vector_type(4)))  float    v4f;

#define S_LEN   2048
#define NBATCH  4
#define NHEADS  16
#define DHEAD   64
#define DMODEL  1024
#define TENSOR_ELEMS (NBATCH * NHEADS * S_LEN * DHEAD)   // 8,388,608

// ---- WMMA fragment helpers (layouts per CDNA5 ISA 7.12.2, wave32) ----------
// A (16x32 f16): lane m = L%16; element e -> k = (e/8)*16 + (L/16)*8 + e%8
__device__ __forceinline__ v16h frag_a(const _Float16* row, int kh) {
  union { v16h v; v8h h[2]; } u;
  u.h[0] = *(const v8h*)(row + kh * 8);
  u.h[1] = *(const v8h*)(row + 16 + kh * 8);
  return u.v;
}
// B (32x16 f16): lane n = L%16; element e -> k = (e/8)*16 + (L/16)*8 + e%8  (row n of [n][k])
__device__ __forceinline__ v16h frag_b(const _Float16* row, int kh) {
  union { v16h v; v8h h[2]; } u;
  u.h[0] = *(const v8h*)(row + kh * 8);
  u.h[1] = *(const v8h*)(row + 16 + kh * 8);
  return u.v;
}
__device__ __forceinline__ v8f wmma16(v16h a, v16h b, v8f c) {
  return __builtin_amdgcn_wmma_f32_16x16x32_f16(false, a, false, b,
                                                (short)0, c, false, false);
}
// convert 16 consecutive f32 -> 16 f16 in LDS (two aligned b128 stores)
__device__ __forceinline__ void cvt16(const float* __restrict__ src,
                                      _Float16* dst) {
  const v4f* s4 = (const v4f*)src;
  v8h o0, o1;
#pragma unroll
  for (int q = 0; q < 2; ++q) {
    v4f x = s4[q];
#pragma unroll
    for (int i = 0; i < 4; ++i) o0[q * 4 + i] = (_Float16)x[i];
  }
#pragma unroll
  for (int q = 0; q < 2; ++q) {
    v4f x = s4[2 + q];
#pragma unroll
    for (int i = 0; i < 4; ++i) o1[q * 4 + i] = (_Float16)x[i];
  }
  *(v8h*)dst = o0;
  *(v8h*)(dst + 8) = o1;
}

// ---------------------------------------------------------------------------
// Kernel 0: RoPE (cos,sin) table. 65536 entries, runs once, lives in L2.
// ---------------------------------------------------------------------------
__global__ __launch_bounds__(32) void rope_table_kernel(float2* __restrict__ cs) {
  const int s = blockIdx.x;        // position
  const int i = threadIdx.x;       // freq index (dd/2)
  const float inv_freq = __expf(-(float)(2 * i) * (9.210340371976184f / 64.f));
  const float ang = (float)s * inv_freq;
  cs[(size_t)s * 32 + i] = make_float2(cosf(ang), sinf(ang));
}

// ---------------------------------------------------------------------------
// Kernel 1: KQV projection + fused RoPE.
// Tile: 256(M) x 64(N), 8 waves; wave w owns rows w*32..w*32+31 (two 16-row
// A-subtiles reused against each B-fragment -> 12 ds_load_b128 per 8 WMMA).
// LDS strides padded (40 halves = 20 dwords, gcd(20,64)=4) -> conflict-free.
// ---------------------------------------------------------------------------
__global__ __launch_bounds__(256) void kqv_rope_kernel(
    const float* __restrict__ X, const float* __restrict__ W,
    const float2* __restrict__ csTab, _Float16* __restrict__ Qh,
    _Float16* __restrict__ Kh, _Float16* __restrict__ Vh) {
  __shared__ __align__(16) _Float16 lA[256 * 40];
  __shared__ __align__(16) _Float16 lB[64 * 40];
  const int tid = threadIdx.x;
  const int lane = tid & 31, wave = tid >> 5;
  const int lg = lane & 15, kh = lane >> 4;
  const int m0 = blockIdx.y * 256;   // token-row tile
  const int n0 = blockIdx.x * 64;    // KQV-column tile

  v8f acc[2][4];
#pragma unroll
  for (int ms = 0; ms < 2; ++ms)
#pragma unroll
    for (int j = 0; j < 4; ++j)
#pragma unroll
      for (int r = 0; r < 8; ++r) acc[ms][j][r] = 0.f;

  for (int k0 = 0; k0 < DMODEL; k0 += 32) {
    {  // stage A tile 256x32 (f32 -> f16), one row per thread
      const float* src = X + (size_t)(m0 + tid) * DMODEL + k0;
      _Float16* dst = lA + tid * 40;
      cvt16(src, dst);
      cvt16(src + 16, dst + 16);
    }
    if (tid < 128) {  // stage B tile 64x32 from W rows (B = W^T)
      int row = tid >> 1, seg = tid & 1;
      cvt16(W + (size_t)(n0 + row) * DMODEL + k0 + seg * 16,
            lB + row * 40 + seg * 16);
    }
    __syncthreads();
    const v16h a0 = frag_a(lA + (wave * 32 + lg) * 40, kh);
    const v16h a1 = frag_a(lA + (wave * 32 + 16 + lg) * 40, kh);
#pragma unroll
    for (int j = 0; j < 4; ++j) {
      const v16h b = frag_b(lB + (j * 16 + lg) * 40, kh);
      acc[0][j] = wmma16(a0, b, acc[0][j]);
      acc[1][j] = wmma16(a1, b, acc[1][j]);
    }
    __syncthreads();
  }

  // Epilogue: split K/Q/V, RoPE via table (pairs are adjacent lanes).
#pragma unroll
  for (int ms = 0; ms < 2; ++ms) {
    const int mw = m0 + wave * 32 + ms * 16;
#pragma unroll
    for (int j = 0; j < 4; ++j) {
      const int n = n0 + j * 16 + lg;
      const int seg = n >> 10;       // 0=K 1=Q 2=V (uniform across wave)
      const int nn = n & 1023;
      const int h = nn >> 6, dd = nn & 63;
      _Float16* outT = (seg == 0) ? Kh : ((seg == 1) ? Qh : Vh);
      const bool doRope = (seg < 2);
      const float2* csRow = csTab + (dd >> 1);
#pragma unroll
      for (int r = 0; r < 8; ++r) {
        const int m = mw + kh * 8 + r;
        const int s = m & (S_LEN - 1), b = m >> 11;
        float v = acc[ms][j][r];
        if (doRope) {
          const float2 cs = csRow[(size_t)s * 32];
          const float p = __shfl_xor(v, 1, 32);
          v = (dd & 1) ? (p * cs.y + v * cs.x) : (v * cs.x - p * cs.y);
        }
        outT[((size_t)(b * NHEADS + h) * S_LEN + s) * DHEAD + dd] = (_Float16)v;
      }
    }
  }
}

// ---------------------------------------------------------------------------
// Kernel 2: causal flash attention, transposed-scores formulation.
// Workgroup = (b,h, 128 q rows); wave w owns q rows w*16..+15.
//   S^T = K_blk(32x64) x Q^T(64x16)  -> lane-local softmax over keys
//   O^T = V^T(64x32)  x P^T(32x16)   -> accumulated per 32-key block
// ---------------------------------------------------------------------------
__global__ __launch_bounds__(256) void attn_kernel(
    const _Float16* __restrict__ Qh, const _Float16* __restrict__ Kh,
    const _Float16* __restrict__ Vh, _Float16* __restrict__ acts) {
  __shared__ __align__(16) _Float16 lK[32 * 72];       // K block [key][d]
  __shared__ __align__(16) _Float16 lV[64 * 40];       // V^T block [d][key]
  __shared__ __align__(16) _Float16 lP[8][16 * 40];    // per-wave P^T [q][key]
  const int tid = threadIdx.x;
  const int lane = tid & 31, wave = tid >> 5;
  const int lg = lane & 15, kh = lane >> 4;
  const int qb = blockIdx.x;
  const int bh = blockIdx.y;                 // b*16 + h
  const int b = bh >> 4, h = bh & 15;
  const _Float16* Qb = Qh + (size_t)bh * S_LEN * DHEAD;
  const _Float16* Kb = Kh + (size_t)bh * S_LEN * DHEAD;
  const _Float16* Vb = Vh + (size_t)bh * S_LEN * DHEAD;
  const int q0 = qb * 128 + wave * 16;
  const int qg = q0 + lg;                    // this lane's query row

  // Q^T B-fragments (live in 16 VGPRs for the whole workgroup lifetime)
  v16h qf0, qf1;
  {
    const _Float16* qrow = Qb + (size_t)qg * DHEAD;
    qf0 = frag_b(qrow, kh);        // d 0..31
    qf1 = frag_b(qrow + 32, kh);   // d 32..63
  }

  v8f ov[4];
#pragma unroll
  for (int c = 0; c < 4; ++c)
#pragma unroll
    for (int r = 0; r < 8; ++r) ov[c][r] = 0.f;
  float mrun = -1e30f, lrun = 0.f;

  const int nblk = qb * 4 + 4;  // causal: only key blocks <= q tile end
  for (int jb = 0; jb < nblk; ++jb) {
    {  // cooperatively stage K block (row-major) and V block (transposed)
      const int kk = tid >> 3, d0 = (tid & 7) * 8;
      const size_t g = (size_t)(jb * 32 + kk) * DHEAD + d0;
      *(v8h*)(lK + kk * 72 + d0) = *(const v8h*)(Kb + g);
      const v8h vv = *(const v8h*)(Vb + g);
#pragma unroll
      for (int i = 0; i < 8; ++i) lV[(d0 + i) * 40 + kk] = vv[i];
    }
    __syncthreads();

    // scores^T: two 16(key) x 16(q) tiles, K-dim = head dim (2 steps of 32)
    v8f st[2];
#pragma unroll
    for (int hb = 0; hb < 2; ++hb) {
#pragma unroll
      for (int r = 0; r < 8; ++r) st[hb][r] = 0.f;
      const _Float16* krow = lK + (hb * 16 + lg) * 72;
      st[hb] = wmma16(frag_a(krow, kh), qf0, st[hb]);
      st[hb] = wmma16(frag_a(krow + 32, kh), qf1, st[hb]);
    }

    // online softmax (per-lane over 16 keys, + one shfl for the other half)
    float p[16];
    float mb = -1e30f;
#pragma unroll
    for (int hb = 0; hb < 2; ++hb)
#pragma unroll
      for (int r = 0; r < 8; ++r) {
        const int key = jb * 32 + hb * 16 + kh * 8 + r;
        float sc = st[hb][r] * 0.125f;      // 1/sqrt(64)
        if (key > qg) sc = -1e30f;          // causal mask
        p[hb * 8 + r] = sc;
        mb = fmaxf(mb, sc);
      }
    mb = fmaxf(mb, __shfl_xor(mb, 16, 32));
    const float mnew = fmaxf(mrun, mb);
    const float alpha = __expf(mrun - mnew);
    float sum = 0.f;
    v8h ph0, ph1;
#pragma unroll
    for (int r = 0; r < 8; ++r) {
      const float e0 = __expf(p[r] - mnew);
      const float e1 = __expf(p[8 + r] - mnew);
      sum += e0 + e1;
      ph0[r] = (_Float16)e0;
      ph1[r] = (_Float16)e1;
    }
    sum += __shfl_xor(sum, 16, 32);
    lrun = lrun * alpha + sum;
    mrun = mnew;
#pragma unroll
    for (int c = 0; c < 4; ++c)
#pragma unroll
      for (int r = 0; r < 8; ++r) ov[c][r] *= alpha;

    // D-layout -> B-layout via per-wave LDS round trip
    _Float16* prow = &lP[wave][lg * 40];
    *(v8h*)(prow + kh * 8) = ph0;        // keys kh*8..+7
    *(v8h*)(prow + 16 + kh * 8) = ph1;   // keys 16+kh*8..+7
    const v16h pb = frag_b(prow, kh);    // P^T as 32x16 B fragment

    // O^T += V^T x P^T, 4 head-dim chunks of 16
#pragma unroll
    for (int c = 0; c < 4; ++c) {
      const v16h va = frag_a(lV + (c * 16 + lg) * 40, kh);
      ov[c] = wmma16(va, pb, ov[c]);
    }
    __syncthreads();
  }

  // normalize and emit acts[b][s][h*64 + dd] (f16 for the output GEMM)
  const float inv = 1.0f / lrun;
  _Float16* obase =
      acts + ((size_t)(b * S_LEN + qg)) * DMODEL + h * DHEAD;
#pragma unroll
  for (int c = 0; c < 4; ++c) {
    v8h o;
#pragma unroll
    for (int r = 0; r < 8; ++r) o[r] = (_Float16)(ov[c][r] * inv);
    *(v8h*)(obase + c * 16 + kh * 8) = o;
  }
}

// ---------------------------------------------------------------------------
// Kernel 3: out = acts @ Wo^T (f16 A, f32->f16 staged B, f32 output).
// Same 256x64 tiling / 32-rows-per-wave as kernel 1.
// ---------------------------------------------------------------------------
__global__ __launch_bounds__(256) void out_proj_kernel(
    const _Float16* __restrict__ A, const float* __restrict__ W,
    float* __restrict__ out) {
  __shared__ __align__(16) _Float16 lA[256 * 40];
  __shared__ __align__(16) _Float16 lB[64 * 40];
  const int tid = threadIdx.x;
  const int lane = tid & 31, wave = tid >> 5;
  const int lg = lane & 15, kh = lane >> 4;
  const int m0 = blockIdx.y * 256;
  const int n0 = blockIdx.x * 64;

  v8f acc[2][4];
#pragma unroll
  for (int ms = 0; ms < 2; ++ms)
#pragma unroll
    for (int j = 0; j < 4; ++j)
#pragma unroll
      for (int r = 0; r < 8; ++r) acc[ms][j][r] = 0.f;

  for (int k0 = 0; k0 < DMODEL; k0 += 32) {
    {  // stage A tile (already f16): four b128 copies per thread
      const _Float16* src = A + (size_t)(m0 + tid) * DMODEL + k0;
      _Float16* dst = lA + tid * 40;
      *(v8h*)dst = *(const v8h*)src;
      *(v8h*)(dst + 8) = *(const v8h*)(src + 8);
      *(v8h*)(dst + 16) = *(const v8h*)(src + 16);
      *(v8h*)(dst + 24) = *(const v8h*)(src + 24);
    }
    if (tid < 128) {
      int row = tid >> 1, seg = tid & 1;
      cvt16(W + (size_t)(n0 + row) * DMODEL + k0 + seg * 16,
            lB + row * 40 + seg * 16);
    }
    __syncthreads();
    const v16h a0 = frag_a(lA + (wave * 32 + lg) * 40, kh);
    const v16h a1 = frag_a(lA + (wave * 32 + 16 + lg) * 40, kh);
#pragma unroll
    for (int j = 0; j < 4; ++j) {
      const v16h b = frag_b(lB + (j * 16 + lg) * 40, kh);
      acc[0][j] = wmma16(a0, b, acc[0][j]);
      acc[1][j] = wmma16(a1, b, acc[1][j]);
    }
    __syncthreads();
  }

#pragma unroll
  for (int ms = 0; ms < 2; ++ms) {
    const int mw = m0 + wave * 32 + ms * 16;
#pragma unroll
    for (int j = 0; j < 4; ++j) {
      const int n = n0 + j * 16 + lg;
#pragma unroll
      for (int r = 0; r < 8; ++r) {
        const int m = mw + kh * 8 + r;
        out[(size_t)m * DMODEL + n] = acc[ms][j][r];
      }
    }
  }
}

// ---------------------------------------------------------------------------
extern "C" void kernel_launch(void* const* d_in, const int* in_sizes, int n_in,
                              void* d_out, int out_size, void* d_ws,
                              size_t ws_size, hipStream_t stream) {
  (void)in_sizes; (void)n_in; (void)out_size; (void)ws_size;
  const float* X    = (const float*)d_in[0];
  const float* Wkqv = (const float*)d_in[1];
  const float* Wo   = (const float*)d_in[2];
  float* out = (float*)d_out;

  _Float16* Qh   = (_Float16*)d_ws;           // each f16 tensor: 16 MB
  _Float16* Kh   = Qh + TENSOR_ELEMS;
  _Float16* Vh   = Kh + TENSOR_ELEMS;
  _Float16* acts = Vh + TENSOR_ELEMS;
  float2*  csTab = (float2*)(acts + TENSOR_ELEMS);  // +512 KB; total ~64.5 MB

  // RoPE cos/sin table: 2048 positions x 32 freqs
  rope_table_kernel<<<dim3(S_LEN), dim3(32), 0, stream>>>(csTab);
  // M=8192 rows, N=3072 cols -> (48, 32) tiles of 64x256
  kqv_rope_kernel<<<dim3(48, 32), dim3(256), 0, stream>>>(X, Wkqv, csTab,
                                                          Qh, Kh, Vh);
  // (q-tile, b*h) = (16, 64)
  attn_kernel<<<dim3(16, 64), dim3(256), 0, stream>>>(Qh, Kh, Vh, acts);
  // M=8192, N=1024 -> (16, 32)
  out_proj_kernel<<<dim3(16, 32), dim3(256), 0, stream>>>(acts, Wo, out);
}
